// SG_51831665328282
// MI455X (gfx1250) — compile-verified, matrix operand order checked by reading, not run on hardware
//
#include <hip/hip_runtime.h>
#include <hip/hip_bf16.h>
#include <cstdint>

typedef __attribute__((ext_vector_type(16))) _Float16 v16h;
typedef __attribute__((ext_vector_type(8)))  float    v8f;

#define NPTS 1024
#define NB   16
#define KNN  20

// ---------------------------------------------------------------------------
// CDNA5 async Global->LDS DMA (ASYNCcnt path, ISA 15.18.3). Copies 16 bytes
// per lane from global f16 panels straight into LDS tiles — no VGPR staging,
// no per-element cvt chains.
// ---------------------------------------------------------------------------
__device__ __forceinline__ void async_g2l(unsigned ldsoff, const _Float16* g) {
    asm volatile("global_load_async_to_lds_b128 %0, %1, off"
                 :: "v"(ldsoff), "v"((unsigned long long)(uintptr_t)g)
                 : "memory");
}
__device__ __forceinline__ void async_wait0() {
    asm volatile("s_wait_asynccnt 0x0" ::: "memory");
}
__device__ __forceinline__ unsigned lds_off(const void* p) {
    return (unsigned)(uintptr_t)p;  // low 32 bits of flat shared ptr = LDS offset
}

// ---------------------------------------------------------------------------
// WMMA fragment helpers (16x16x32 f16 -> f32).
// A 16x32: lane&15 = row M, lane>>4 selects K-half; VGPR j holds K pairs.
// B 32x16: lane&15 = col N, same K mapping along rows.
// C/D 16x16: lane&15 = col N, VGPR r = row r + 8*(lane>>4).
// ---------------------------------------------------------------------------
__device__ __forceinline__ int frag_kbase(int j, int h) {
    return (j < 4) ? (h * 8 + 2 * j) : (16 + h * 8 + 2 * (j - 4));
}

__device__ __forceinline__ v16h lds_load_Afrag(const _Float16* As, int ld) {
    int lane = threadIdx.x & 31, m = lane & 15, h = lane >> 4;
    v16h a;
#pragma unroll
    for (int j = 0; j < 8; ++j) {
        int kb = frag_kbase(j, h);
        a[2 * j]     = As[m * ld + kb];
        a[2 * j + 1] = As[m * ld + kb + 1];
    }
    return a;
}

// A-fragment from a K-major (transposed) LDS tile: tile[k][m], ld in m units.
__device__ __forceinline__ v16h lds_load_AfragT(const _Float16* As, int ld) {
    int lane = threadIdx.x & 31, m = lane & 15, h = lane >> 4;
    v16h a;
#pragma unroll
    for (int j = 0; j < 8; ++j) {
        int kb = frag_kbase(j, h);
        a[2 * j]     = As[kb * ld + m];
        a[2 * j + 1] = As[(kb + 1) * ld + m];
    }
    return a;
}

__device__ __forceinline__ v16h lds_load_Bfrag(const _Float16* Bs, int ld) {
    int lane = threadIdx.x & 31, n = lane & 15, h = lane >> 4;
    v16h b;
#pragma unroll
    for (int j = 0; j < 8; ++j) {
        int kb = frag_kbase(j, h);
        b[2 * j]     = Bs[kb * ld + n];
        b[2 * j + 1] = Bs[(kb + 1) * ld + n];
    }
    return b;
}

// ---------------------------------------------------------------------------
// Weight f32 -> zero-padded f16 panel: dst[r*kp + c] = (c<ncols)?src[r*lda+aoff+c]:0
// ---------------------------------------------------------------------------
__global__ void k_wcvt(const float* __restrict__ src, int lda, int aoff,
                       int ncols, int kp, _Float16* __restrict__ dst)
{
    long i = (long)blockIdx.x * 256 + threadIdx.x;
    int c = (int)(i % kp);
    int r = (int)(i / kp);
    dst[i] = (c < ncols) ? (_Float16)src[(long)r * lda + aoff + c] : (_Float16)0.0f;
}

// ---------------------------------------------------------------------------
// Generic WMMA GEMM on f16 operands: Out[b] = A16[b](MxKp) * B16[b](KpxN)
// (+Res f32) (+=Out). Kp multiple of 32, tiles staged via async LDS DMA.
// ---------------------------------------------------------------------------
__global__ void __launch_bounds__(32) k_gemm(
    const _Float16* __restrict__ A16, int lda, long sAb,
    const _Float16* __restrict__ B16, int ldb, long sBb,
    float* __restrict__ Out, int ldo, long sOb,
    int Kp, const float* __restrict__ Res, long sRb, int accumulate)
{
    __shared__ _Float16 As[16 * 32];
    __shared__ _Float16 Bs[32 * 16];
    int row0 = blockIdx.x * 16, col0 = blockIdx.y * 16, b = blockIdx.z;
    const _Float16* Ab = A16 + (long)b * sAb;
    const _Float16* Bb = B16 + (long)b * sBb;
    float* Ob = Out + (long)b * sOb;
    int tid = threadIdx.x;
    unsigned ldsA = lds_off(As), ldsB = lds_off(Bs);
    v8f acc = {};
    int ktn = Kp >> 5;
    for (int kt = 0; kt < ktn; ++kt) {
        int k0 = kt << 5;
        // A tile 16x32 halves (64B/row): 64 x 16B units, 2 per lane
#pragma unroll
        for (int ph = 0; ph < 2; ++ph) {
            int u = tid + 32 * ph, row = u >> 2, ch = u & 3;
            async_g2l(ldsA + row * 64 + ch * 16,
                      Ab + (long)(row0 + row) * lda + k0 + ch * 8);
        }
        // B tile 32x16 halves (32B/row): 64 x 16B units, 2 per lane
#pragma unroll
        for (int ph = 0; ph < 2; ++ph) {
            int u = tid + 32 * ph, row = u >> 1, ch = u & 1;
            async_g2l(ldsB + row * 32 + ch * 16,
                      Bb + (long)(k0 + row) * ldb + col0 + ch * 8);
        }
        if (kt + 1 < ktn)  // speculative prefetch of the next B panel
            __builtin_prefetch(Bb + (long)(k0 + 32 + (tid >> 1)) * ldb + col0, 0, 0);
        async_wait0();
        __syncthreads();
        v16h af = lds_load_Afrag(As, 32);
        v16h bf = lds_load_Bfrag(Bs, 16);
        acc = __builtin_amdgcn_wmma_f32_16x16x32_f16(false, af, false, bf,
                                                     (short)0, acc, false, false);
        __syncthreads();
    }
    int n = tid & 15, h = tid >> 4;
#pragma unroll
    for (int r = 0; r < 8; ++r) {
        int m = r + 8 * h;
        long o = (long)(row0 + m) * ldo + col0 + n;
        float v = acc[r];
        if (Res) v += Res[(long)b * sRb + o];
        if (accumulate) v += Ob[o];
        Ob[o] = v;
    }
}

// ---------------------------------------------------------------------------
// Input transpose: (B,N,200)[:,:,choff:choff+100] -> X f32 + f16 mirror
// (mirror zero-padded to 128 channels so Kp=128 GEMM/KNN is exact).
// ---------------------------------------------------------------------------
__global__ void k_tin(const float* __restrict__ src, float* __restrict__ X,
                      _Float16* __restrict__ X16, int choff)
{
    long i = (long)blockIdx.x * 256 + threadIdx.x;  // B*N
    int n = (int)(i % NPTS), b = (int)(i / NPTS);
    for (int c = 0; c < 100; ++c) {
        float v = src[((long)b * NPTS + n) * 200 + choff + c];
        X[((long)b * 256 + c) * NPTS + n] = v;
        X16[((long)b * 256 + c) * NPTS + n] = (_Float16)v;
    }
    for (int c = 100; c < 128; ++c)
        X16[((long)b * 256 + c) * NPTS + n] = (_Float16)0.0f;
}

__global__ void k_norms(const float* __restrict__ X, float* __restrict__ nrm, int C)
{
    long i = (long)blockIdx.x * 256 + threadIdx.x;
    int n = (int)(i % NPTS), b = (int)(i / NPTS);
    float s = 0.f;
    for (int c = 0; c < C; ++c) {
        float v = X[((long)b * 256 + c) * NPTS + n];
        s += v * v;
    }
    nrm[b * NPTS + n] = s;
}

// ---------------------------------------------------------------------------
// KNN: Gram tiles via WMMA from f16 mirrors; A tile staged K-major (points
// are the contiguous direction) so the async DMA needs no transpose.
// ---------------------------------------------------------------------------
__global__ void __launch_bounds__(32) k_knn(
    const _Float16* __restrict__ X16, int Kp, const float* __restrict__ nrm,
    int* __restrict__ idx)
{
    __shared__ _Float16 Aall[128 * 16];   // [channel][query]
    __shared__ _Float16 Bs[32 * 16];      // [channel][point]
    __shared__ float dtile[16 * 16];
    __shared__ float bestv[16 * KNN];
    __shared__ int   besti[16 * KNN];
    int q0 = blockIdx.x * 16, b = blockIdx.y;
    int tid = threadIdx.x, n = tid & 15, h = tid >> 4;
    const _Float16* Xb = X16 + (long)b * 256 * NPTS;
    unsigned ldsA = lds_off(Aall), ldsB = lds_off(Bs);
    // stage all A chunks: 128 rows x 32B -> 256 units
#pragma unroll
    for (int ph = 0; ph < 8; ++ph) {
        int u = tid + 32 * ph, row = u >> 1, ch = u & 1;
        async_g2l(ldsA + row * 32 + ch * 16, Xb + (long)row * NPTS + q0 + ch * 8);
    }
    if (tid < 16)
        for (int kk = 0; kk < KNN; ++kk) { bestv[tid * KNN + kk] = -3.0e38f; besti[tid * KNN + kk] = 0; }
    async_wait0();
    __syncthreads();
    int chunks = Kp >> 5;
    v16h afr[4];
#pragma unroll
    for (int kt = 0; kt < 4; ++kt) afr[kt] = lds_load_AfragT(Aall + kt * 32 * 16, 16);
    float qn[8];
#pragma unroll
    for (int r = 0; r < 8; ++r) qn[r] = nrm[b * NPTS + q0 + r + 8 * h];

    for (int m0 = 0; m0 < NPTS; m0 += 16) {
        v8f acc = {};
        for (int kt = 0; kt < chunks; ++kt) {
#pragma unroll
            for (int ph = 0; ph < 2; ++ph) {
                int u = tid + 32 * ph, row = u >> 1, ch = u & 1;
                async_g2l(ldsB + row * 32 + ch * 16,
                          Xb + (long)((kt << 5) + row) * NPTS + m0 + ch * 8);
            }
            async_wait0();
            __syncthreads();
            v16h bf = lds_load_Bfrag(Bs, 16);
            acc = __builtin_amdgcn_wmma_f32_16x16x32_f16(false, afr[kt], false, bf,
                                                         (short)0, acc, false, false);
            __syncthreads();
        }
        float cn = nrm[b * NPTS + m0 + n];
#pragma unroll
        for (int r = 0; r < 8; ++r) {
            int m = r + 8 * h;
            dtile[m * 16 + n] = 2.0f * acc[r] - qn[r] - cn;
        }
        __syncthreads();
        if (tid < 16) {
            int row = tid;
            float mn = bestv[row * KNN]; int mnp = 0;
            for (int kk = 1; kk < KNN; ++kk)
                if (bestv[row * KNN + kk] < mn) { mn = bestv[row * KNN + kk]; mnp = kk; }
            for (int j = 0; j < 16; ++j) {
                float v = dtile[row * 16 + j];
                if (v > mn) {
                    bestv[row * KNN + mnp] = v; besti[row * KNN + mnp] = m0 + j;
                    mn = bestv[row * KNN]; mnp = 0;
                    for (int kk = 1; kk < KNN; ++kk)
                        if (bestv[row * KNN + kk] < mn) { mn = bestv[row * KNN + kk]; mnp = kk; }
                }
            }
        }
        __syncthreads();
    }
    if (tid < 16)
        for (int kk = 0; kk < KNN; ++kk)
            idx[((long)b * NPTS + q0 + tid) * KNN + kk] = besti[tid * KNN + kk];
}

// ---------------------------------------------------------------------------
// Edge aggregation + BN machinery (unchanged math)
// ---------------------------------------------------------------------------
__global__ void __launch_bounds__(256) k_edge_agg(
    const float* __restrict__ G1, const float* __restrict__ G2,
    const int* __restrict__ idx, float* __restrict__ ymax,
    float* __restrict__ ymin, float* __restrict__ stats)
{
    __shared__ float rs[256], rs2[256];
    int tid = threadIdx.x;
    int b = blockIdx.z, o = blockIdx.y, n = blockIdx.x * 256 + tid;
    const float* g1 = G1 + ((long)b * 256 + o) * NPTS;
    float g1n = g1[n];
    float base = G2[((long)b * 256 + o) * NPTS + n] - g1n;
    const int* id = idx + ((long)b * NPTS + n) * KNN;
    float mx = -3.0e38f, mn = 3.0e38f, s = 0.f, s2 = 0.f;
    for (int k = 0; k < KNN; ++k) {
        float y = g1[id[k]] + base;
        mx = fmaxf(mx, y); mn = fminf(mn, y); s += y; s2 += y * y;
    }
    long off = ((long)b * 256 + o) * NPTS + n;
    ymax[off] = mx; ymin[off] = mn;
    rs[tid] = s; rs2[tid] = s2; __syncthreads();
    for (int st = 128; st > 0; st >>= 1) {
        if (tid < st) { rs[tid] += rs[tid + st]; rs2[tid] += rs2[tid + st]; }
        __syncthreads();
    }
    if (tid == 0) { atomicAdd(&stats[o * 2], rs[0]); atomicAdd(&stats[o * 2 + 1], rs2[0]); }
}

__global__ void k_bn_fin(const float* __restrict__ stats, const float* __restrict__ g,
                         const float* __restrict__ bb, float* __restrict__ scsh,
                         int co, float cnt)
{
    int o = threadIdx.x;
    if (o >= co) return;
    float mean = stats[o * 2] / cnt;
    float var = stats[o * 2 + 1] / cnt - mean * mean;
    float sc = g[o] * rsqrtf(var + 1e-5f);
    scsh[o * 2] = sc;
    scsh[o * 2 + 1] = bb[o] - mean * sc;
}

__global__ void k_edge_apply(const float* __restrict__ ymax, const float* __restrict__ ymin,
                             const float* __restrict__ scsh, float* __restrict__ xout,
                             _Float16* __restrict__ xout16, int co)
{
    long i = (long)blockIdx.x * 256 + threadIdx.x;  // B*co*N
    int n = (int)(i % NPTS); long t = i / NPTS;
    int o = (int)(t % co), b = (int)(t / co);
    float sc = scsh[o * 2], sh = scsh[o * 2 + 1];
    long off = ((long)b * 256 + o) * NPTS + n;
    float v = (sc >= 0.f ? ymax[off] : ymin[off]) * sc + sh;
    v = v >= 0.f ? v : 0.2f * v;
    xout[off] = v;
    xout16[off] = (_Float16)v;
}

__global__ void __launch_bounds__(256) k_bn1_stats(const float* __restrict__ G,
                                                   float* __restrict__ stats)
{
    __shared__ float rs[256], rs2[256];
    int tid = threadIdx.x;
    int b = blockIdx.z, o = blockIdx.y, n = blockIdx.x * 256 + tid;
    float y = G[((long)b * 256 + o) * NPTS + n];
    rs[tid] = y; rs2[tid] = y * y; __syncthreads();
    for (int st = 128; st > 0; st >>= 1) {
        if (tid < st) { rs[tid] += rs[tid + st]; rs2[tid] += rs2[tid + st]; }
        __syncthreads();
    }
    if (tid == 0) { atomicAdd(&stats[o * 2], rs[0]); atomicAdd(&stats[o * 2 + 1], rs2[0]); }
}

__global__ void k_bn1_apply(const float* __restrict__ G, const float* __restrict__ scsh,
                            float* __restrict__ a, _Float16* __restrict__ a16)
{
    long i = (long)blockIdx.x * 256 + threadIdx.x;  // B*256*N
    int n = (int)(i % NPTS); long t = i / NPTS;
    int o = (int)(t % 256), b = (int)(t / 256);
    float v = G[((long)b * 256 + o) * NPTS + n] * scsh[o * 2] + scsh[o * 2 + 1];
    v = v >= 0.f ? v : 0.2f * v;
    long to = ((long)b * NPTS + n) * 256 + o;  // token-major
    a[to] = v;
    a16[to] = (_Float16)v;
}

// ---------------------------------------------------------------------------
// Flash attention (dh=32): QK^T + 2 padded P@V WMMAs per key tile, online
// softmax via 16-lane shfl_xor reductions. Emits f16 mirror of O.
// ---------------------------------------------------------------------------
__global__ void __launch_bounds__(32) k_flash(
    const float* __restrict__ Q, const float* __restrict__ Kmat,
    const float* __restrict__ V, _Float16* __restrict__ O16)
{
    __shared__ _Float16 Ps[16 * 32];
    int q0 = blockIdx.x * 16;
    int bh = blockIdx.y, b = bh >> 3, hh = bh & 7;
    int tid = threadIdx.x, n = tid & 15, half = tid >> 4;
    const float scale = 0.17677669529663687f;  // 1/sqrt(32)
    long base = (long)b * NPTS * 256 + hh * 32;

    v16h aq;
#pragma unroll
    for (int j = 0; j < 8; ++j) {
        int kb = frag_kbase(j, half);
        long t = base + (long)(q0 + n) * 256;
        aq[2 * j]     = (_Float16)Q[t + kb];
        aq[2 * j + 1] = (_Float16)Q[t + kb + 1];
    }
    for (int i = tid; i < 16 * 32; i += 32) Ps[i] = (_Float16)0.0f;
    float mrun[8], lrun[8];
    v8f acc0 = {}, acc1 = {};
#pragma unroll
    for (int r = 0; r < 8; ++r) { mrun[r] = -3.0e38f; lrun[r] = 0.0f; }
    __syncthreads();

    for (int m0 = 0; m0 < NPTS; m0 += 16) {
        v16h bk;
#pragma unroll
        for (int j = 0; j < 8; ++j) {
            int kb = frag_kbase(j, half);
            long t = base + (long)(m0 + n) * 256;
            bk[2 * j]     = (_Float16)Kmat[t + kb];
            bk[2 * j + 1] = (_Float16)Kmat[t + kb + 1];
        }
        v8f zc = {};
        v8f s = __builtin_amdgcn_wmma_f32_16x16x32_f16(false, aq, false, bk,
                                                       (short)0, zc, false, false);
        float pr[8];
#pragma unroll
        for (int r = 0; r < 8; ++r) {
            float v = s[r] * scale;
            float tm = v;
            for (int off = 1; off < 16; off <<= 1) tm = fmaxf(tm, __shfl_xor(tm, off, 32));
            float mnew = fmaxf(mrun[r], tm);
            float alpha = __expf(mrun[r] - mnew);
            float p = __expf(v - mnew);
            float rsum = p;
            for (int off = 1; off < 16; off <<= 1) rsum += __shfl_xor(rsum, off, 32);
            lrun[r] = lrun[r] * alpha + rsum;
            mrun[r] = mnew;
            acc0[r] *= alpha; acc1[r] *= alpha;
            pr[r] = p;
        }
#pragma unroll
        for (int r = 0; r < 8; ++r) Ps[(r + 8 * half) * 32 + n] = (_Float16)pr[r];
        __syncthreads();
        v16h ap = lds_load_Afrag(Ps, 32);
        v16h bv0, bv1;
#pragma unroll
        for (int j = 0; j < 8; ++j) {
            int kb = frag_kbase(j, half);
            _Float16 x0a = 0, x0b = 0, x1a = 0, x1b = 0;
            if (kb < 16) {
                long t0 = base + (long)(m0 + kb) * 256;
                x0a = (_Float16)V[t0 + n];
                x1a = (_Float16)V[t0 + 16 + n];
            }
            if (kb + 1 < 16) {
                long t1 = base + (long)(m0 + kb + 1) * 256;
                x0b = (_Float16)V[t1 + n];
                x1b = (_Float16)V[t1 + 16 + n];
            }
            bv0[2 * j] = x0a; bv0[2 * j + 1] = x0b;
            bv1[2 * j] = x1a; bv1[2 * j + 1] = x1b;
        }
        acc0 = __builtin_amdgcn_wmma_f32_16x16x32_f16(false, ap, false, bv0,
                                                      (short)0, acc0, false, false);
        acc1 = __builtin_amdgcn_wmma_f32_16x16x32_f16(false, ap, false, bv1,
                                                      (short)0, acc1, false, false);
        __syncthreads();
    }
#pragma unroll
    for (int r = 0; r < 8; ++r) {
        int m = r + 8 * half;
        float inv = 1.0f / lrun[r];
        long t = base + (long)(q0 + m) * 256;
        O16[t + n]      = (_Float16)(acc0[r] * inv);
        O16[t + 16 + n] = (_Float16)(acc1[r] * inv);
    }
}

// ---------------------------------------------------------------------------
// Row LayerNorm over 256 features; emits f32 + f16 mirror.
// ---------------------------------------------------------------------------
__global__ void __launch_bounds__(256) k_ln(const float* __restrict__ X,
                                            const float* __restrict__ g,
                                            const float* __restrict__ bb,
                                            float* __restrict__ Y,
                                            _Float16* __restrict__ Y16)
{
    __shared__ float rs[256], rs2[256];
    int t = blockIdx.x, f = threadIdx.x;
    float v = X[(long)t * 256 + f];
    rs[f] = v; rs2[f] = v * v; __syncthreads();
    for (int st = 128; st > 0; st >>= 1) {
        if (f < st) { rs[f] += rs[f + st]; rs2[f] += rs2[f + st]; }
        __syncthreads();
    }
    float mean = rs[0] * (1.0f / 256.0f);
    float var = rs2[0] * (1.0f / 256.0f) - mean * mean;
    float y = (v - mean) * rsqrtf(var + 1e-5f) * g[f] + bb[f];
    Y[(long)t * 256 + f] = y;
    Y16[(long)t * 256 + f] = (_Float16)y;
}

// ---------------------------------------------------------------------------
// GeM pooling + head
// ---------------------------------------------------------------------------
__global__ void __launch_bounds__(256) k_gem(const float* __restrict__ A,
                                             const float* __restrict__ pptr,
                                             float* __restrict__ e)
{
    int b = blockIdx.x, f = threadIdx.x;
    float p = *pptr;
    float s = 0.f;
    for (int nn = 0; nn < NPTS; ++nn) {
        float v = fmaxf(A[((long)b * NPTS + nn) * 256 + f], 1e-6f);
        s += __powf(v, p);
    }
    e[b * 256 + f] = __powf(s * (1.0f / NPTS), 1.0f / p);
}

__global__ void __launch_bounds__(256) k_head_t(const float* __restrict__ e1,
                                                const float* __restrict__ e2,
                                                const float* __restrict__ W,
                                                const float* __restrict__ Wb,
                                                const float* __restrict__ tb,
                                                float* __restrict__ t)
{
    __shared__ float rs[256];
    int b = blockIdx.x >> 4, i = blockIdx.x & 15, f = threadIdx.x;
    const float* e2b = e2 + b * 256;
    float inner = 0.f;
    for (int gix = 0; gix < 256; ++gix)
        inner += W[((long)f * 256 + gix) * 16 + i] * e2b[gix];
    rs[f] = e1[b * 256 + f] * inner;
    __syncthreads();
    for (int st = 128; st > 0; st >>= 1) {
        if (f < st) rs[f] += rs[f + st];
        __syncthreads();
    }
    if (f == 0) {
        float acc = rs[0] + tb[i];
        for (int j = 0; j < 256; ++j) acc += e1[b * 256 + j] * Wb[i * 512 + j];
        for (int j = 0; j < 256; ++j) acc += e2[b * 256 + j] * Wb[i * 512 + 256 + j];
        t[b * 16 + i] = fmaxf(acc, 0.f);
    }
}

__global__ void k_head_fin(const float* __restrict__ t, const float* __restrict__ fcW,
                           const float* __restrict__ fcb, const float* __restrict__ scW,
                           const float* __restrict__ scb, float* __restrict__ out)
{
    int b = threadIdx.x;
    if (b < NB) {
        float s = scb[0];
        for (int i = 0; i < 16; ++i) {
            float h = fcb[i];
            for (int j = 0; j < 16; ++j) h += t[b * 16 + j] * fcW[i * 16 + j];
            h = fmaxf(h, 0.f);
            s += h * scW[i];
        }
        out[b] = 1.0f / (1.0f + __expf(-s));
    }
}

// ---------------------------------------------------------------------------
// Host orchestration
// ---------------------------------------------------------------------------
extern "C" void kernel_launch(void* const* d_in, const int* in_sizes, int n_in,
                              void* d_out, int out_size, void* d_ws, size_t ws_size,
                              hipStream_t stream)
{
    (void)in_sizes; (void)n_in; (void)out_size; (void)ws_size;
    const size_t SLOT = (size_t)NB * 256 * NPTS * sizeof(float);  // 16.78 MB
    const size_t HSLOT = SLOT / 2;
    char* ws = (char*)d_ws;
    auto slot = [&](int i) { return (float*)(ws + (size_t)i * SLOT); };
    char* hbase = ws + 10 * SLOT;
    auto hslot = [&](int i) { return (_Float16*)(hbase + (size_t)i * HSLOT); };
    size_t off = 10 * SLOT + 10 * HSLOT;
    auto bump = [&](size_t bytes) { size_t o = off; off += (bytes + 255) & ~(size_t)255; return o; };
    int* idxbuf = (int*)(ws + bump((size_t)NB * NPTS * KNN * sizeof(int)));
    float* norms = (float*)(ws + bump((size_t)NB * NPTS * sizeof(float)));
    float* stats = (float*)(ws + bump(2 * 256 * sizeof(float)));
    float* scsh = (float*)(ws + bump(2 * 256 * sizeof(float)));
    float* e1 = (float*)(ws + bump(NB * 256 * sizeof(float)));
    float* e2 = (float*)(ws + bump(NB * 256 * sizeof(float)));
    float* tvec = (float*)(ws + bump(NB * 16 * sizeof(float)));
    _Float16* whp = (_Float16*)(ws + bump(4u << 20));  // 4MB weight-panel pool
    auto walloc = [&](size_t nh) { _Float16* p = whp; whp += nh; return p; };

    const long CSTRIDE = 256L * NPTS;   // f32 channel-major batch stride
    const long HSTRIDE = 256L * NPTS;   // f16 mirror batch stride (halves)
    float* G1 = slot(2);
    float* G2 = slot(3);
    float* ymax = slot(4);
    float* ymin = slot(5);

    // ---- one-time weight f16 panel conversions (zero-padded K) ----
    int Cs[3] = {100, 64, 128}, cos[3] = {64, 128, 256}, kps[3] = {128, 64, 128};
    _Float16* whL[2][3][2];
    for (int br = 0; br < 2; ++br)
        for (int l = 0; l < 3; ++l) {
            const float* W = (const float*)d_in[(br ? 11 : 2) + 3 * l];
            for (int hf = 0; hf < 2; ++hf) {
                whL[br][l][hf] = walloc((size_t)cos[l] * kps[l]);
                k_wcvt<<<cos[l] * kps[l] / 256, 256, 0, stream>>>(
                    W, 2 * Cs[l], hf * Cs[l], Cs[l], kps[l], whL[br][l][hf]);
            }
        }
    _Float16* whE[2];
    for (int hf = 0; hf < 2; ++hf) {
        whE[hf] = walloc(256 * 256);
        k_wcvt<<<256 * 256 / 256, 256, 0, stream>>>(
            (const float*)d_in[20], 512, hf * 256, 256, 256, whE[hf]);
    }
    _Float16* whA[2][4];
    for (int blk = 0; blk < 2; ++blk)
        for (int w = 0; w < 4; ++w) {
            whA[blk][w] = walloc(256 * 256);
            k_wcvt<<<256 * 256 / 256, 256, 0, stream>>>(
                (const float*)d_in[23 + 6 * blk + w], 256, 0, 256, 256, whA[blk][w]);
        }

    // ---- edge-conv passes ----
    auto edge_pass = [&](const float* inFeat, float* aOut, _Float16* aOut16) {
        struct XB { float* f; _Float16* h; };
        XB bufs[2][2] = {{{slot(0), hslot(0)}, {slot(1), hslot(1)}},
                         {{slot(6), hslot(6)}, {slot(7), hslot(7)}}};
        int pgbase[2] = {0, 1};
        (void)pgbase;
        int choffs[2] = {0, 100};
        XB bout[2];
        for (int br = 0; br < 2; ++br) {
            XB xcur = bufs[br][0], xnext = bufs[br][1];
            k_tin<<<NB * NPTS / 256, 256, 0, stream>>>(inFeat, xcur.f, xcur.h, choffs[br]);
            int C = 100;
            for (int l = 0; l < 3; ++l) {
                const float* gg = (const float*)d_in[(br ? 11 : 2) + 3 * l + 1];
                const float* bb = (const float*)d_in[(br ? 11 : 2) + 3 * l + 2];
                int co = cos[l], kp = kps[l];
                k_norms<<<NB * NPTS / 256, 256, 0, stream>>>(xcur.f, norms, C);
                k_knn<<<dim3(NPTS / 16, NB), 32, 0, stream>>>(xcur.h, kp, norms, idxbuf);
                k_gemm<<<dim3(co / 16, NPTS / 16, NB), 32, 0, stream>>>(
                    whL[br][l][0], kp, 0, xcur.h, NPTS, HSTRIDE,
                    G1, NPTS, CSTRIDE, kp, nullptr, 0, 0);
                k_gemm<<<dim3(co / 16, NPTS / 16, NB), 32, 0, stream>>>(
                    whL[br][l][1], kp, 0, xcur.h, NPTS, HSTRIDE,
                    G2, NPTS, CSTRIDE, kp, nullptr, 0, 0);
                hipMemsetAsync(stats, 0, 2 * 256 * sizeof(float), stream);
                k_edge_agg<<<dim3(NPTS / 256, co, NB), 256, 0, stream>>>(
                    G1, G2, idxbuf, ymax, ymin, stats);
                k_bn_fin<<<1, 256, 0, stream>>>(stats, gg, bb, scsh, co,
                                                (float)(NB * NPTS * KNN));
                k_edge_apply<<<NB * co * NPTS / 256, 256, 0, stream>>>(
                    ymax, ymin, scsh, xnext.f, xnext.h, co);
                XB tmp = xcur; xcur = xnext; xnext = tmp;
                C = co;
            }
            bout[br] = xcur;
        }
        k_gemm<<<dim3(16, NPTS / 16, NB), 32, 0, stream>>>(
            whE[0], 256, 0, bout[0].h, NPTS, HSTRIDE,
            G1, NPTS, CSTRIDE, 256, nullptr, 0, 0);
        k_gemm<<<dim3(16, NPTS / 16, NB), 32, 0, stream>>>(
            whE[1], 256, 0, bout[1].h, NPTS, HSTRIDE,
            G1, NPTS, CSTRIDE, 256, nullptr, 0, 1);
        hipMemsetAsync(stats, 0, 2 * 256 * sizeof(float), stream);
        k_bn1_stats<<<dim3(NPTS / 256, 256, NB), 256, 0, stream>>>(G1, stats);
        k_bn_fin<<<1, 256, 0, stream>>>(stats, (const float*)d_in[21],
                                        (const float*)d_in[22], scsh, 256,
                                        (float)(NB * NPTS));
        k_bn1_apply<<<NB * 256 * NPTS / 256, 256, 0, stream>>>(G1, scsh, aOut, aOut16);
    };

    edge_pass((const float*)d_in[0], slot(8), hslot(8));  // a1
    edge_pass((const float*)d_in[1], slot(9), hslot(9));  // a2

    // ---- cross-attention blocks ----
    const int T = NB * NPTS;
    auto attn_block = [&](int blk, const float* x1f, const _Float16* x1h,
                          const float* x2f, const _Float16* x2h,
                          float* y1f, _Float16* y1h, float* y2f, _Float16* y2h) {
        const float* lg = (const float*)d_in[23 + 6 * blk + 4];
        const float* lb = (const float*)d_in[23 + 6 * blk + 5];
        float *Qb = slot(0), *Kb = slot(1), *Vb = slot(2);
        _Float16* Oh = hslot(3);
        auto mha = [&](const float* qf, const _Float16* qh, const _Float16* kvh,
                       float* yf, _Float16* yh) {
            k_gemm<<<dim3(T / 16, 16, 1), 32, 0, stream>>>(
                qh, 256, 0, whA[blk][0], 256, 0, Qb, 256, 0, 256, nullptr, 0, 0);
            k_gemm<<<dim3(T / 16, 16, 1), 32, 0, stream>>>(
                kvh, 256, 0, whA[blk][1], 256, 0, Kb, 256, 0, 256, nullptr, 0, 0);
            k_gemm<<<dim3(T / 16, 16, 1), 32, 0, stream>>>(
                kvh, 256, 0, whA[blk][2], 256, 0, Vb, 256, 0, 256, nullptr, 0, 0);
            k_flash<<<dim3(NPTS / 16, NB * 8), 32, 0, stream>>>(Qb, Kb, Vb, Oh);
            // pre = O @ Wo + residual(q_in) into Kb (free after flash)
            k_gemm<<<dim3(T / 16, 16, 1), 32, 0, stream>>>(
                Oh, 256, 0, whA[blk][3], 256, 0, Kb, 256, 0, 256, qf, 0, 0);
            k_ln<<<T, 256, 0, stream>>>(Kb, lg, lb, yf, yh);
        };
        mha(x1f, x1h, x2h, y1f, y1h);
        mha(x2f, x2h, x1h, y2f, y2h);
    };

    attn_block(0, slot(8), hslot(8), slot(9), hslot(9),
               slot(4), hslot(4), slot(5), hslot(5));
    attn_block(1, slot(4), hslot(4), slot(5), hslot(5),
               slot(8), hslot(8), slot(9), hslot(9));

    // ---- head ----
    const float* gp = (const float*)d_in[35];
    k_gem<<<NB, 256, 0, stream>>>(slot(8), gp, e1);
    k_gem<<<NB, 256, 0, stream>>>(slot(9), gp, e2);
    k_head_t<<<NB * 16, 256, 0, stream>>>(e1, e2, (const float*)d_in[36],
                                          (const float*)d_in[37],
                                          (const float*)d_in[38], tvec);
    k_head_fin<<<1, 32, 0, stream>>>(tvec, (const float*)d_in[39],
                                     (const float*)d_in[40],
                                     (const float*)d_in[41],
                                     (const float*)d_in[42], (float*)d_out);
}
